// SPRGraphNet_88648124990053
// MI455X (gfx1250) — compile-verified, matrix operand order checked by reading, not run on hardware
//
#include <hip/hip_runtime.h>

#define N_NODES  50000
#define N_EDGES  800000
#define N_GRAPHS 512
#define EMB      128
#define HID      256      // == 2*EMB, so both layers have K = 256
#define NCLS     32

typedef __attribute__((ext_vector_type(16))) _Float16 v16h;
typedef __attribute__((ext_vector_type(8)))  float    v8f;

// ---------------------------------------------------------------------------
// WMMA fragment helpers (16-bit 16x32 A layout per CDNA5 ISA 7.12.2):
//   lane&15  -> matrix row (A: M, B: output column N via W row)
//   lane>>4  -> K-half (adds 8)
//   VGPR v   -> K pair at (v&3)*2 + (v>>2)*16
// ---------------------------------------------------------------------------
__device__ __forceinline__ v16h load_a_lds(const _Float16* base, int stride,
                                           int lane, int kbase) {
  const int m = lane & 15;
  const int kh = (lane >> 4) << 3;
  const _Float16* rp = base + m * stride + kbase + kh;
  v16h f;
#pragma unroll
  for (int v = 0; v < 8; ++v) {
    const int k0 = ((v & 3) << 1) + ((v >> 2) << 4);
    f[2 * v]     = rp[k0];
    f[2 * v + 1] = rp[k0 + 1];
  }
  return f;
}

// B[k][n] = W[n][k]; W pre-converted to f16, row-major [rows][ldw]
// Per lane this is two contiguous 16B runs -> two global_load_b128, no VALU.
__device__ __forceinline__ v16h load_b_h(const _Float16* __restrict__ W, int ldw,
                                         int nbase, int lane, int kbase) {
  const int n = nbase + (lane & 15);
  const int kh = (lane >> 4) << 3;
  const _Float16* rp = W + (size_t)n * ldw + kbase + kh;
  v16h f;
#pragma unroll
  for (int v = 0; v < 8; ++v) {
    const int k0 = ((v & 3) << 1) + ((v >> 2) << 4);
    f[2 * v]     = rp[k0];
    f[2 * v + 1] = rp[k0 + 1];
  }
  return f;
}

// ---------------------------------------------------------------------------
// Utility kernels
// ---------------------------------------------------------------------------
__global__ void zero_kernel(float* __restrict__ p, int n) {
  int i = blockIdx.x * blockDim.x + threadIdx.x;
  if (i < n) p[i] = 0.0f;
}

__global__ void cvt_f16_kernel(const float* __restrict__ s,
                               _Float16* __restrict__ d, int n) {
  int i = blockIdx.x * blockDim.x + threadIdx.x;
  if (i < n) d[i] = (_Float16)s[i];
}

// h0[i][c] = c<128 ? shape_emb[x[i,0]][c] : color_emb[x[i,1]][c-128]
__global__ void embed_kernel(const int* __restrict__ x,
                             const float* __restrict__ se,
                             const float* __restrict__ ce,
                             float* __restrict__ h0) {
  int idx = blockIdx.x * blockDim.x + threadIdx.x;   // over N*256
  int i = idx >> 8;
  int c = idx & 255;
  if (i >= N_NODES) return;
  float v;
  if (c < EMB) v = se[x[2 * i] * EMB + c];
  else         v = ce[x[2 * i + 1] * EMB + (c - EMB)];
  h0[idx] = v;
}

__global__ void deg_kernel(const int* __restrict__ dst, float* __restrict__ deg) {
  int e = blockIdx.x * blockDim.x + threadIdx.x;
  if (e < N_EDGES) atomicAdd(&deg[dst[e]], 1.0f);
}

__global__ void cnt_kernel(const int* __restrict__ batch, float* __restrict__ cnt) {
  int i = blockIdx.x * blockDim.x + threadIdx.x;
  if (i < N_NODES) atomicAdd(&cnt[batch[i]], 1.0f);
}

// one wave32 per edge; float4 gathers (global_load_b128) + per-f32 atomics.
// agg (51 MB) is L2-resident on MI455X (192 MB L2), so atomics resolve in L2.
__global__ void scatter_kernel(const int* __restrict__ src,
                               const int* __restrict__ dst,
                               const float* __restrict__ feat,
                               float* __restrict__ agg) {
  int gid = blockIdx.x * blockDim.x + threadIdx.x;
  int e = gid >> 5;
  int lane = gid & 31;
  if (e >= N_EDGES) return;
  const float4* fp = (const float4*)(feat + (size_t)src[e] * HID);
  float* ap = agg + (size_t)dst[e] * HID;
#pragma unroll
  for (int j = 0; j < 2; ++j) {
    int q = lane + (j << 5);           // float4 index 0..63
    float4 v = fp[q];
    float* a = ap + (q << 2);
    atomicAdd(a + 0, v.x);
    atomicAdd(a + 1, v.y);
    atomicAdd(a + 2, v.z);
    atomicAdd(a + 3, v.w);
  }
}

__global__ void pool_kernel(const int* __restrict__ batch,
                            const float* __restrict__ h,
                            float* __restrict__ pooled) {
  int gid = blockIdx.x * blockDim.x + threadIdx.x;
  int i = gid >> 5;
  int lane = gid & 31;
  if (i >= N_NODES) return;
  const float4* hp = (const float4*)(h + (size_t)i * HID);
  float* pp = pooled + (size_t)batch[i] * HID;
#pragma unroll
  for (int j = 0; j < 2; ++j) {
    int q = lane + (j << 5);
    float4 v = hp[q];
    float* p = pp + (q << 2);
    atomicAdd(p + 0, v.x);
    atomicAdd(p + 1, v.y);
    atomicAdd(p + 2, v.z);
    atomicAdd(p + 3, v.w);
  }
}

// ---------------------------------------------------------------------------
// SAGE layer dense part: hout = relu( (agg/deg) @ wl^T + bl + hin @ wr^T )
// Block = 512 threads (16 waves), owns 16 node rows; wave w -> cols [16w,16w+16)
// ---------------------------------------------------------------------------
__global__ __launch_bounds__(512) void sage_mm_kernel(
    const float* __restrict__ hin, const float* __restrict__ agg,
    const float* __restrict__ deg,
    const _Float16* __restrict__ wl, const float* __restrict__ bl,
    const _Float16* __restrict__ wr, float* __restrict__ hout) {
  __shared__ _Float16 sAgg[16][HID + 8];
  __shared__ _Float16 sH[16][HID + 8];

  const int tid = threadIdx.x;
  const int mbase = blockIdx.x << 4;

  // stage 16 rows x 256 cols of both operands, fuse the degree-mean
#pragma unroll
  for (int it = 0; it < 8; ++it) {
    int idx = tid + (it << 9);
    int r = idx >> 8;
    int c = idx & 255;
    int node = mbase + r;
    float sc = 1.0f / fmaxf(deg[node], 1.0f);
    sAgg[r][c] = (_Float16)(agg[(size_t)node * HID + c] * sc);
    sH[r][c]   = (_Float16)(hin[(size_t)node * HID + c]);
  }
  __syncthreads();

  const int wave = tid >> 5;
  const int lane = tid & 31;
  const int nbase = wave << 4;

  v8f acc = {};
#pragma unroll
  for (int kk = 0; kk < HID; kk += 32) {
    v16h a = load_a_lds(&sAgg[0][0], HID + 8, lane, kk);
    v16h b = load_b_h(wl, HID, nbase, lane, kk);
    acc = __builtin_amdgcn_wmma_f32_16x16x32_f16(false, a, false, b,
                                                 (short)0, acc, false, false);
  }
#pragma unroll
  for (int kk = 0; kk < HID; kk += 32) {
    v16h a = load_a_lds(&sH[0][0], HID + 8, lane, kk);
    v16h b = load_b_h(wr, HID, nbase, lane, kk);
    acc = __builtin_amdgcn_wmma_f32_16x16x32_f16(false, a, false, b,
                                                 (short)0, acc, false, false);
  }

  // C/D layout: lane&15 -> N, lane>>4 selects M+8, VGPR r -> M row
  const int n = nbase + (lane & 15);
  const float bias = bl[n];
  const int mo = (lane >> 4) << 3;
#pragma unroll
  for (int r = 0; r < 8; ++r) {
    int m = mbase + mo + r;
    float v = acc[r] + bias;
    hout[(size_t)m * HID + n] = fmaxf(v, 0.0f);
  }
}

// ---------------------------------------------------------------------------
// Classifier: out[g][n] = (pooled[g]/cnt[g]) . wc[n] + bc[n]; 64 WMMA tiles
// ---------------------------------------------------------------------------
__global__ __launch_bounds__(512) void classifier_kernel(
    const float* __restrict__ pooled, const float* __restrict__ cnt,
    const _Float16* __restrict__ wc, const float* __restrict__ bc,
    float* __restrict__ out) {
  const int tid = threadIdx.x;
  const int wave = tid >> 5;
  const int lane = tid & 31;
  const int t = (blockIdx.x << 4) + wave;   // 0..63 tiles
  const int mbase = (t >> 1) << 4;          // graph-row tile
  const int nbase = (t & 1) << 4;           // class-col tile

  const int m = mbase + (lane & 15);
  const float ic = 1.0f / fmaxf(cnt[m], 1.0f);
  const int kh = (lane >> 4) << 3;

  v8f acc = {};
#pragma unroll
  for (int kk = 0; kk < HID; kk += 32) {
    const float* rp = pooled + (size_t)m * HID + kk + kh;
    v16h a;
#pragma unroll
    for (int v = 0; v < 8; ++v) {
      const int k0 = ((v & 3) << 1) + ((v >> 2) << 4);
      a[2 * v]     = (_Float16)(rp[k0] * ic);
      a[2 * v + 1] = (_Float16)(rp[k0 + 1] * ic);
    }
    v16h b = load_b_h(wc, HID, nbase, lane, kk);
    acc = __builtin_amdgcn_wmma_f32_16x16x32_f16(false, a, false, b,
                                                 (short)0, acc, false, false);
  }

  const int n = nbase + (lane & 15);
  const float bias = bc[n];
  const int mo = (lane >> 4) << 3;
#pragma unroll
  for (int r = 0; r < 8; ++r) {
    int g = mbase + mo + r;
    out[(size_t)g * NCLS + n] = acc[r] + bias;
  }
}

// ---------------------------------------------------------------------------
// Host-side orchestration
// ---------------------------------------------------------------------------
extern "C" void kernel_launch(void* const* d_in, const int* in_sizes, int n_in,
                              void* d_out, int out_size, void* d_ws, size_t ws_size,
                              hipStream_t stream) {
  (void)in_sizes; (void)n_in; (void)out_size; (void)ws_size;

  const int*   x     = (const int*)d_in[0];
  const int*   ei    = (const int*)d_in[1];
  const int*   batch = (const int*)d_in[2];
  const float* se    = (const float*)d_in[3];
  const float* ce    = (const float*)d_in[4];
  const float* w1l   = (const float*)d_in[5];
  const float* b1l   = (const float*)d_in[6];
  const float* w1r   = (const float*)d_in[7];
  const float* w2l   = (const float*)d_in[8];
  const float* b2l   = (const float*)d_in[9];
  const float* w2r   = (const float*)d_in[10];
  const float* wc    = (const float*)d_in[11];
  const float* bc    = (const float*)d_in[12];
  float* out = (float*)d_out;

  const size_t NF = (size_t)N_NODES * HID;      // 12.8M floats
  float* ws     = (float*)d_ws;
  float* h0     = ws;                              // [N,256]
  float* agg    = ws + NF;                         // [N,256] (both layers)
  float* h1     = ws + 2 * NF;                     // [N,256]
  float* deg    = ws + 3 * NF;                     // [N]
  float* pooled = deg + N_NODES;                   // [G,256]
  float* cnt    = pooled + (size_t)N_GRAPHS * HID; // [G]
  float* h2     = h0;                              // reuse h0 after layer 1

  // f16 weight copies (after cnt; ~520KB)
  _Float16* w1l_h = (_Float16*)(cnt + N_GRAPHS);
  _Float16* w1r_h = w1l_h + HID * HID;
  _Float16* w2l_h = w1r_h + HID * HID;
  _Float16* w2r_h = w2l_h + HID * HID;
  _Float16* wc_h  = w2r_h + HID * HID;

  const int* src = ei;
  const int* dst = ei + N_EDGES;

  const int NFi = (int)NF;
  const int WN  = HID * HID;                       // 65536

  // init accumulators
  zero_kernel<<<(NFi + 255) / 256, 256, 0, stream>>>(agg, NFi);
  zero_kernel<<<(N_NODES + 255) / 256, 256, 0, stream>>>(deg, N_NODES);
  zero_kernel<<<(N_GRAPHS * HID + 255) / 256, 256, 0, stream>>>(pooled, N_GRAPHS * HID);
  zero_kernel<<<(N_GRAPHS + 255) / 256, 256, 0, stream>>>(cnt, N_GRAPHS);

  // one-shot f16 weight conversion
  cvt_f16_kernel<<<WN / 256, 256, 0, stream>>>(w1l, w1l_h, WN);
  cvt_f16_kernel<<<WN / 256, 256, 0, stream>>>(w1r, w1r_h, WN);
  cvt_f16_kernel<<<WN / 256, 256, 0, stream>>>(w2l, w2l_h, WN);
  cvt_f16_kernel<<<WN / 256, 256, 0, stream>>>(w2r, w2r_h, WN);
  cvt_f16_kernel<<<(NCLS * HID) / 256, 256, 0, stream>>>(wc, wc_h, NCLS * HID);

  // embeddings, degrees, graph sizes
  embed_kernel<<<(NFi + 255) / 256, 256, 0, stream>>>(x, se, ce, h0);
  deg_kernel<<<(N_EDGES + 255) / 256, 256, 0, stream>>>(dst, deg);
  cnt_kernel<<<(N_NODES + 255) / 256, 256, 0, stream>>>(batch, cnt);

  // layer 1
  scatter_kernel<<<N_EDGES / 8, 256, 0, stream>>>(src, dst, h0, agg);
  sage_mm_kernel<<<N_NODES / 16, 512, 0, stream>>>(h0, agg, deg, w1l_h, b1l, w1r_h, h1);

  // layer 2
  zero_kernel<<<(NFi + 255) / 256, 256, 0, stream>>>(agg, NFi);
  scatter_kernel<<<N_EDGES / 8, 256, 0, stream>>>(src, dst, h1, agg);
  sage_mm_kernel<<<N_NODES / 16, 512, 0, stream>>>(h1, agg, deg, w2l_h, b2l, w2r_h, h2);

  // global mean pool + classifier
  pool_kernel<<<N_NODES * 32 / 256, 256, 0, stream>>>(batch, h2, pooled);
  classifier_kernel<<<4, 512, 0, stream>>>(pooled, cnt, wc_h, bc, out);
}
